// PolygonTransformerPerceiver_9809705304975
// MI455X (gfx1250) — matrix operand order repaired and checked by reading, not compiled, not run on hardware
//
#include <hip/hip_runtime.h>
#include <hip/hip_bf16.h>
#include <math.h>

// ---------------------------------------------------------------------------
// PolygonTransformerPerceiver forward, MI455X (gfx1250, wave32, WMMA bf16).
// All dense GEMMs: v_wmma_f32_16x16x32_bf16, 128x64 block tile, 32x32 register
// tile per wave (4 WMMA / step), K templated (32/256/1024) so there is no
// tail code. A-tile staged via global_load_async_to_lds_b128 (ASYNCcnt).
// Edge softmax via atomics; Perceiver attention wave-per-row.
// ---------------------------------------------------------------------------

#define TRY_ASYNC_LDS 1   // set 0 if the integrated assembler rejects the op

typedef __attribute__((ext_vector_type(16))) __bf16        v16bf;
typedef __attribute__((ext_vector_type(8)))  float         v8f;
typedef __attribute__((ext_vector_type(4)))  unsigned int  u32x4;

#define NN   32768      // nodes
#define NE   262144     // edges
#define NB   64         // graphs
#define PP   512        // nodes per graph
#define NLAT 64
#define NHEAD 8

__device__ __forceinline__ float gelu_erf(float v) {
  return 0.5f * v * (1.0f + erff(v * 0.7071067811865475f));
}

// ======================= WMMA GEMM =========================================
// C[M,Nc] = A[M,K] @ W[K,Nc]  (A,W bf16 row-major). M % 128 == 0, Nc % 64 == 0,
// K compile-time (multiple of 32).
// MODE 0: outf = r   MODE 1: outf += r   MODE 2: outb = bf16(r)
template<int MODE, bool BIAS, bool GELU, int K>
__global__ __launch_bounds__(256) void k_gemm(const __bf16* __restrict__ A,
                                              const __bf16* __restrict__ Bw,
                                              const float*  __restrict__ bias,
                                              float*        __restrict__ outf,
                                              __bf16*       __restrict__ outb,
                                              int Nc) {
  __shared__ __bf16 As [128][40];  // 128 rows x BK=32 (+8 pad)
  __shared__ __bf16 BsT[64][40];   // 64 cols  x BK=32 (+8 pad), transposed B
  const int t    = threadIdx.x;
  const int w    = t >> 5, lane = t & 31;
  const int wm   = w >> 1, wn   = w & 1;     // 4x2 waves -> 128x64 tile
  const int rowBase = blockIdx.y * 128;
  const int colBase = blockIdx.x * 64;
  const int arow = t >> 1, acg = (t & 1) * 16;   // A loader: 16 elems/thread
  const int bkr  = t >> 3, bcg = (t & 7) * 8;    // B loader: 8 elems/thread

  const __bf16* aPtr = A  + (size_t)(rowBase + arow) * K + acg;
  const __bf16* bPtr = Bw + (size_t)bkr * Nc + colBase + bcg;

  v8f accs[2][2];
  #pragma unroll
  for (int i = 0; i < 2; ++i)
    #pragma unroll
    for (int j = 0; j < 2; ++j)
      accs[i][j] = (v8f){0.f,0.f,0.f,0.f,0.f,0.f,0.f,0.f};

  #pragma unroll 1
  for (int k0 = 0; k0 < K; k0 += 32) {
#if TRY_ASYNC_LDS
    { // async DMA global -> LDS for the A tile (tracked by ASYNCcnt)
      unsigned lA = (unsigned)(uintptr_t)&As[arow][acg];
      asm volatile("global_load_async_to_lds_b128 %0, %1, off\n\t"
                   "global_load_async_to_lds_b128 %2, %3, off"
                   :: "v"(lA), "v"(aPtr + k0), "v"(lA + 16u), "v"(aPtr + k0 + 8)
                   : "memory");
    }
#else
    *(u32x4*)&As[arow][acg]     = *(const u32x4*)(aPtr + k0);
    *(u32x4*)&As[arow][acg + 8] = *(const u32x4*)(aPtr + k0 + 8);
#endif
    { // B tile, stored transposed: BsT[n][k]
      union { u32x4 u; __bf16 e[8]; } bb;
      bb.u = *(const u32x4*)(bPtr + (size_t)k0 * Nc);
      #pragma unroll
      for (int j = 0; j < 8; ++j) BsT[bcg + j][bkr] = bb.e[j];
    }
    if (k0 + 32 < K) __builtin_prefetch(bPtr + (size_t)(k0 + 32) * Nc, 0, 1);
#if TRY_ASYNC_LDS
    asm volatile("s_wait_asynccnt 0x0" ::: "memory");
#endif
    __syncthreads();

    // Fragment loads per ISA 7.12.2 (wave32).
    union Frag { v16bf v; u32x4 q[2]; };
    Frag fa0, fa1, fb0, fb1;
    const int fm  = lane & 15;
    const int akb = (lane < 16) ? 0 : 8;    // A: lanes 16-31 start at K=8
    const int am0 = (wm << 5) + fm;
    fa0.q[0] = *(const u32x4*)&As[am0][akb];
    fa0.q[1] = *(const u32x4*)&As[am0][akb + 16];
    fa1.q[0] = *(const u32x4*)&As[am0 + 16][akb];
    fa1.q[1] = *(const u32x4*)&As[am0 + 16][akb + 16];
    const int bn  = (wn << 5) + fm;
    const int bkb = (lane < 16) ? 0 : 8;    // B: lanes 16-31 start at K=8
    fb0.q[0] = *(const u32x4*)&BsT[bn][bkb];
    fb0.q[1] = *(const u32x4*)&BsT[bn][bkb + 16];
    fb1.q[0] = *(const u32x4*)&BsT[bn + 16][bkb];
    fb1.q[1] = *(const u32x4*)&BsT[bn + 16][bkb + 16];

    accs[0][0] = __builtin_amdgcn_wmma_f32_16x16x32_bf16(false, fa0.v, false, fb0.v,
                                                         (short)0, accs[0][0], false, false);
    accs[0][1] = __builtin_amdgcn_wmma_f32_16x16x32_bf16(false, fa0.v, false, fb1.v,
                                                         (short)0, accs[0][1], false, false);
    accs[1][0] = __builtin_amdgcn_wmma_f32_16x16x32_bf16(false, fa1.v, false, fb0.v,
                                                         (short)0, accs[1][0], false, false);
    accs[1][1] = __builtin_amdgcn_wmma_f32_16x16x32_bf16(false, fa1.v, false, fb1.v,
                                                         (short)0, accs[1][1], false, false);
    __syncthreads();
  }

  // Epilogue. C/D layout: lane n=lane&15, VGPR v -> row v + (lane>=16 ? 8 : 0).
  const int nE = lane & 15;
  #pragma unroll
  for (int ri = 0; ri < 2; ++ri) {
    const int rBase = rowBase + (wm << 5) + (ri << 4) + ((lane < 16) ? 0 : 8);
    #pragma unroll
    for (int f = 0; f < 2; ++f) {
      v8f acc = accs[ri][f];
      int col = colBase + (wn << 5) + (f << 4) + nE;
      float bv = 0.0f;
      if (BIAS) bv = bias[col];
      #pragma unroll
      for (int vv = 0; vv < 8; ++vv) {
        float r = acc[vv] + bv;
        if (GELU) r = gelu_erf(r);
        size_t o = (size_t)(rBase + vv) * Nc + col;
        if (MODE == 0)      outf[o] = r;
        else if (MODE == 1) outf[o] += r;
        else                outb[o] = (__bf16)r;
      }
    }
  }
}

// ======================= LayerNorm (D=256, wave per row) ===================
template<bool WF32>
__global__ __launch_bounds__(256) void k_ln(const float* __restrict__ x,
                                            const float* __restrict__ gg,
                                            const float* __restrict__ bb,
                                            float*       __restrict__ of,
                                            __bf16*      __restrict__ ob) {
  const int w = threadIdx.x >> 5, lane = threadIdx.x & 31;
  const size_t row = (size_t)blockIdx.x * 8 + w;
  const float* xr = x + row * 256;
  float vals[8]; float s = 0.f;
  #pragma unroll
  for (int j = 0; j < 8; ++j) { vals[j] = xr[lane * 8 + j]; s += vals[j]; }
  #pragma unroll
  for (int m = 16; m >= 1; m >>= 1) s += __shfl_xor(s, m, 32);
  const float mean = s * (1.0f / 256.0f);
  float vs = 0.f;
  #pragma unroll
  for (int j = 0; j < 8; ++j) { float d = vals[j] - mean; vs += d * d; }
  #pragma unroll
  for (int m = 16; m >= 1; m >>= 1) vs += __shfl_xor(vs, m, 32);
  const float rstd = rsqrtf(vs * (1.0f / 256.0f) + 1e-5f);
  #pragma unroll
  for (int j = 0; j < 8; ++j) {
    int c = lane * 8 + j;
    float r = (vals[j] - mean) * rstd * gg[c] + bb[c];
    if (WF32) of[row * 256 + c] = r;
    ob[row * 256 + c] = (__bf16)r;
  }
}

// ======================= utility kernels ===================================
__global__ void k_cvt(const float* __restrict__ s, __bf16* __restrict__ d, size_t n) {
  size_t i = (size_t)blockIdx.x * blockDim.x + threadIdx.x;
  size_t st = (size_t)gridDim.x * blockDim.x;
  for (; i < n; i += st) d[i] = (__bf16)s[i];
}
// dst[i] = i < ncopy ? bf16(src[i]) : 0   (pads 16x256 weights to 32x256)
__global__ void k_cvt_pad(const float* __restrict__ s, __bf16* __restrict__ d,
                          size_t ncopy, size_t ntot) {
  size_t i = (size_t)blockIdx.x * blockDim.x + threadIdx.x;
  size_t st = (size_t)gridDim.x * blockDim.x;
  for (; i < ntot; i += st) d[i] = (i < ncopy) ? (__bf16)s[i] : (__bf16)0.0f;
}
// rows x 16 f32 -> rows x 32 bf16 (zero upper 16 cols)
__global__ void k_pad16to32(const float* __restrict__ s, __bf16* __restrict__ d,
                            size_t ntot) {
  size_t i = (size_t)blockIdx.x * blockDim.x + threadIdx.x;
  size_t st = (size_t)gridDim.x * blockDim.x;
  for (; i < ntot; i += st) {
    int col = (int)(i & 31);
    d[i] = (col < 16) ? (__bf16)s[(i >> 5) * 16 + col] : (__bf16)0.0f;
  }
}
__global__ void k_fill(unsigned int* __restrict__ p, unsigned int v, size_t n) {
  size_t i = (size_t)blockIdx.x * blockDim.x + threadIdx.x;
  size_t st = (size_t)gridDim.x * blockDim.x;
  for (; i < n; i += st) p[i] = v;
}
__global__ void k_bcast_lat(const float* __restrict__ src, float* __restrict__ lat,
                            __bf16* __restrict__ latb, size_t n) {
  size_t i = (size_t)blockIdx.x * blockDim.x + threadIdx.x;
  size_t st = (size_t)gridDim.x * blockDim.x;
  for (; i < n; i += st) { float v = src[i & 16383]; lat[i] = v; latb[i] = (__bf16)v; }
}
__global__ void k_mean_lat(const float* __restrict__ lat, float* __restrict__ g) {
  int b = blockIdx.x, d = threadIdx.x;
  float s = 0.f;
  for (int l = 0; l < NLAT; ++l) s += lat[((size_t)(b * NLAT + l)) * 256 + d];
  g[b * 256 + d] = s * (1.0f / (float)NLAT);
}
__global__ void k_norm_out(const float* __restrict__ in, float* __restrict__ out) {
  __shared__ float red[256];
  int r = blockIdx.x, t = threadIdx.x;
  float v = in[r * 256 + t];
  red[t] = v * v; __syncthreads();
  for (int s = 128; s > 0; s >>= 1) { if (t < s) red[t] += red[t + s]; __syncthreads(); }
  out[r * 256 + t] = v / fmaxf(sqrtf(red[0]), 1e-12f);
}

// ======================= edge attention (TransformerConv) ==================
__global__ __launch_bounds__(256) void k_edge_alpha(const float* __restrict__ q,
                                                    const float* __restrict__ k,
                                                    const __bf16* __restrict__ ep,
                                                    const int* __restrict__ src,
                                                    const int* __restrict__ dst,
                                                    float* __restrict__ alpha) {
  const int w = threadIdx.x >> 5, lane = threadIdx.x & 31;
  const int e = blockIdx.x * 8 + w;
  if (e >= NE) return;
  const int sN = src[e], dN = dst[e];
  const int d0 = lane * 8;
  const float*  qr = q  + (size_t)dN * 256 + d0;
  const float*  kr = k  + (size_t)sN * 256 + d0;
  const __bf16* er = ep + (size_t)e  * 256 + d0;
  float p = 0.f;
  #pragma unroll
  for (int j = 0; j < 8; ++j) p += qr[j] * (kr[j] + (float)er[j]);
  p += __shfl_xor(p, 1, 32);
  p += __shfl_xor(p, 2, 32);            // reduce 4 lanes of one head
  if ((lane & 3) == 0)
    alpha[(size_t)e * 8 + (lane >> 2)] = p * 0.17677669529663689f;
}
__global__ void k_segmax(const float* __restrict__ alpha, const int* __restrict__ dst,
                         float* __restrict__ m) {
  size_t i = (size_t)blockIdx.x * blockDim.x + threadIdx.x;
  if (i >= (size_t)NE * 8) return;
  int e = (int)(i >> 3), h = (int)(i & 7);
  float v = alpha[i];
  int* a = (int*)&m[(size_t)dst[e] * 8 + h];
  if (v >= 0.f) atomicMax(a, __float_as_int(v));
  else          atomicMin((unsigned int*)a, __float_as_uint(v));
}
__global__ void k_expsum(float* __restrict__ alpha, const int* __restrict__ dst,
                         const float* __restrict__ m, float* __restrict__ s) {
  size_t i = (size_t)blockIdx.x * blockDim.x + threadIdx.x;
  if (i >= (size_t)NE * 8) return;
  int e = (int)(i >> 3), h = (int)(i & 7);
  float ex = __expf(alpha[i] - m[(size_t)dst[e] * 8 + h]);
  alpha[i] = ex;
  atomicAdd(&s[(size_t)dst[e] * 8 + h], ex);
}
__global__ __launch_bounds__(256) void k_edge_scatter(const float* __restrict__ v,
                                                      const __bf16* __restrict__ ep,
                                                      const float* __restrict__ alpha,
                                                      const float* __restrict__ s,
                                                      const int* __restrict__ src,
                                                      const int* __restrict__ dst,
                                                      float* __restrict__ hout) {
  const int w = threadIdx.x >> 5, lane = threadIdx.x & 31;
  const int e = blockIdx.x * 8 + w;
  if (e >= NE) return;
  const int sN = src[e], dN = dst[e];
  const int h = lane >> 2;
  const float a = alpha[(size_t)e * 8 + h] / fmaxf(s[(size_t)dN * 8 + h], 1e-16f);
  const int d0 = lane * 8;
  const float*  vr = v  + (size_t)sN * 256 + d0;
  const __bf16* er = ep + (size_t)e  * 256 + d0;
  float* ho = hout + (size_t)dN * 256 + d0;
  #pragma unroll
  for (int j = 0; j < 8; ++j)
    atomicAdd(&ho[j], (vr[j] + (float)er[j]) * a);
}

// ======================= dense softmax attention (Perceiver) ===============
__global__ __launch_bounds__(256) void k_attn(const float* __restrict__ q,
                                              const float* __restrict__ k,
                                              const float* __restrict__ v,
                                              float* __restrict__ out,
                                              int Lq, int Lk) {
  __shared__ float sc[8][512];
  const int b = blockIdx.x >> 3, h = blockIdx.x & 7;
  const int w = threadIdx.x >> 5, lane = threadIdx.x & 31;
  const float scale = 0.17677669529663689f;
  for (int i = w; i < Lq; i += 8) {
    const float* qr = q + ((size_t)(b * Lq + i) * 256) + h * 32;
    float qv[32];
    #pragma unroll
    for (int d = 0; d < 32; ++d) qv[d] = qr[d];
    float lmax = -1e30f;
    for (int kk = lane; kk < Lk; kk += 32) {
      const float* kr = k + ((size_t)(b * Lk + kk) * 256) + h * 32;
      float dot = 0.f;
      #pragma unroll
      for (int d = 0; d < 32; ++d) dot += qv[d] * kr[d];
      dot *= scale;
      sc[w][kk] = dot;
      lmax = fmaxf(lmax, dot);
    }
    #pragma unroll
    for (int m = 16; m >= 1; m >>= 1) lmax = fmaxf(lmax, __shfl_xor(lmax, m, 32));
    float lsum = 0.f;
    for (int kk = lane; kk < Lk; kk += 32) {
      float ex = __expf(sc[w][kk] - lmax);
      sc[w][kk] = ex;
      lsum += ex;
    }
    #pragma unroll
    for (int m = 16; m >= 1; m >>= 1) lsum += __shfl_xor(lsum, m, 32);
    const float inv = 1.0f / lsum;
    __builtin_amdgcn_wave_barrier();
    float accv = 0.f;
    for (int kk = 0; kk < Lk; ++kk)
      accv += sc[w][kk] * v[((size_t)(b * Lk + kk) * 256) + h * 32 + lane];
    out[((size_t)(b * Lq + i) * 256) + h * 32 + lane] = accv * inv;
    __builtin_amdgcn_wave_barrier();
  }
}

// ======================= host orchestration ================================
#define GEMM(MODE, BIAS, GELU, KV, A, W, BIASP, OF, OB, M, NC) \
  k_gemm<MODE, BIAS, GELU, KV><<<dim3((NC)/64, (M)/128), 256, 0, stream>>>(A, W, BIASP, OF, OB, NC)

extern "C" void kernel_launch(void* const* d_in, const int* in_sizes, int n_in,
                              void* d_out, int out_size, void* d_ws, size_t ws_size,
                              hipStream_t stream) {
  (void)in_sizes; (void)n_in; (void)out_size; (void)ws_size;
  // -------- unpack inputs (setup_inputs() insertion order, params depth-first
  // in construction order from _make_params) --------
  int idx = 0;
  auto nf = [&]() { return (const float*)d_in[idx++]; };
  const float* x_in   = nf();
  const float* eig_in = nf();
  const int*   eidx   = (const int*)d_in[idx++];
  const float* eattr  = nf();
  const int*   batch  = (const int*)d_in[idx++]; (void)batch;
  struct Lin { const float* W; const float* b; };
  auto getL = [&]() { Lin l; l.W = nf(); l.b = nf(); return l; };
  Lin inproj = getL();
  Lin pe1 = getL(), pe2 = getL();
  struct Blk { const float *n1g,*n1b; Lin q,k,v; const float* We; Lin skip;
               const float *n2g,*n2b; Lin ff1, ff2; };
  Blk blk[4];
  for (int L = 0; L < 4; ++L) {
    blk[L].n1g = nf(); blk[L].n1b = nf();
    blk[L].q = getL(); blk[L].k = getL(); blk[L].v = getL();
    blk[L].We = nf();
    blk[L].skip = getL();
    blk[L].n2g = nf(); blk[L].n2b = nf();
    blk[L].ff1 = getL(); blk[L].ff2 = getL();
  }
  const float* latents = nf();
  Lin caq = getL(), cak = getL(), cav = getL(), cao = getL();
  const float *cng = nf(), *cnb = nf();
  Lin saq = getL(), sak = getL(), sav = getL(), sao = getL();
  const float *sng = nf(), *snb = nf();
  Lin pff1 = getL(), pff2 = getL();
  const float *fng = nf(), *fnb = nf();
  const float *olng = nf(), *olnb = nf();
  Lin out1 = getL(), out2 = getL();

  // -------- workspace carve --------
  char* wsp = (char*)d_ws; size_t off = 0;
  auto alloc = [&](size_t bytes) -> void* {
    void* p = wsp + off; off = (off + bytes + 255) & ~(size_t)255; return p;
  };
  float*  hf    = (float*) alloc((size_t)NN * 256 * 4);   // master activations
  __bf16* xb    = (__bf16*)alloc((size_t)NN * 256 * 2);   // bf16 staging
  float*  qf    = (float*) alloc((size_t)NN * 256 * 4);
  float*  kf    = (float*) alloc((size_t)NN * 256 * 4);   // also k_ca
  float*  vf    = (float*) alloc((size_t)NN * 256 * 4);   // also v_ca
  float*  ksaf  = (float*) alloc((size_t)4096 * 256 * 4);
  float*  vsaf  = (float*) alloc((size_t)4096 * 256 * 4);
  __bf16* ebuf  = (__bf16*)alloc((size_t)NE * 256 * 2);   // edge projections
  float*  alpha = (float*) alloc((size_t)NE * 8 * 4);
  float*  mbuf  = (float*) alloc((size_t)NN * 8 * 4);
  float*  sbuf  = (float*) alloc((size_t)NN * 8 * 4);
  __bf16* ffh   = (__bf16*)alloc((size_t)NN * 1024 * 2);  // FFN hidden
  float*  lat   = (float*) alloc((size_t)4096 * 256 * 4);
  __bf16* latb  = (__bf16*)alloc((size_t)4096 * 256 * 2);
  float*  attnf = (float*) alloc((size_t)4096 * 256 * 4);
  __bf16* attnb = (__bf16*)alloc((size_t)4096 * 256 * 2);
  __bf16* xsm   = (__bf16*)alloc((size_t)NN * 32 * 2);    // x padded to K=32
  __bf16* esm   = (__bf16*)alloc((size_t)NN * 32 * 2);    // eig padded to K=32
  __bf16* eab   = (__bf16*)alloc((size_t)NE * 32 * 2);
  float*  gmean = (float*) alloc(64 * 256 * 4);
  __bf16* gbP   = (__bf16*)alloc(128 * 256 * 2);          // zero-padded to M=128
  __bf16* o1b   = (__bf16*)alloc(128 * 256 * 2);
  float*  obuf  = (float*) alloc(128 * 256 * 4);

  auto cvt = [&](const float* s, size_t n) -> __bf16* {
    __bf16* d2 = (__bf16*)alloc(n * 2);
    int blocks = (int)((n + 255) / 256); if (blocks > 4096) blocks = 4096;
    k_cvt<<<blocks, 256, 0, stream>>>(s, d2, n);
    return d2;
  };
  auto cvt_into = [&](const float* s, __bf16* d2, size_t n) {
    int blocks = (int)((n + 255) / 256); if (blocks > 4096) blocks = 4096;
    k_cvt<<<blocks, 256, 0, stream>>>(s, d2, n);
  };
  auto cvt_pad = [&](const float* s, size_t ncopy, size_t ntot) -> __bf16* {
    __bf16* d2 = (__bf16*)alloc(ntot * 2);
    k_cvt_pad<<<(int)((ntot + 255) / 256), 256, 0, stream>>>(s, d2, ncopy, ntot);
    return d2;
  };
  auto fill = [&](void* p, unsigned int v, size_t n) {
    k_fill<<<2048, 256, 0, stream>>>((unsigned int*)p, v, n);
  };

  // -------- bf16 weight conversions (L2-resident, ~9 MB) --------
  __bf16* WiB  = cvt_pad(inproj.W, 16 * 256, 32 * 256);   // pad K 16 -> 32
  __bf16* Wp1  = cvt_pad(pe1.W,    16 * 256, 32 * 256);
  __bf16* Wp2  = cvt(pe2.W, 256 * 256);
  __bf16 *WqB[4], *WkB[4], *WvB[4], *WeB[4], *WsB[4], *Wf1B[4], *Wf2B[4];
  for (int L = 0; L < 4; ++L) {
    WqB[L]  = cvt(blk[L].q.W, 256 * 256);
    WkB[L]  = cvt(blk[L].k.W, 256 * 256);
    WvB[L]  = cvt(blk[L].v.W, 256 * 256);
    WeB[L]  = cvt(blk[L].We, 32 * 256);
    WsB[L]  = cvt(blk[L].skip.W, 256 * 256);
    Wf1B[L] = cvt(blk[L].ff1.W, 256 * 1024);
    Wf2B[L] = cvt(blk[L].ff2.W, 1024 * 256);
  }
  __bf16* WcaQ = cvt(caq.W, 256*256); __bf16* WcaK = cvt(cak.W, 256*256);
  __bf16* WcaV = cvt(cav.W, 256*256); __bf16* WcaO = cvt(cao.W, 256*256);
  __bf16* WsaQ = cvt(saq.W, 256*256); __bf16* WsaK = cvt(sak.W, 256*256);
  __bf16* WsaV = cvt(sav.W, 256*256); __bf16* WsaO = cvt(sao.W, 256*256);
  __bf16* Wpf1 = cvt(pff1.W, 256*1024); __bf16* Wpf2 = cvt(pff2.W, 1024*256);
  __bf16* Wo1  = cvt(out1.W, 256*256);  __bf16* Wo2  = cvt(out2.W, 256*256);

  const int* esrc = eidx;        // edge_index[0]
  const int* edst = eidx + NE;   // edge_index[1]

  // -------- input projection + Laplacian PE encoder --------
  k_pad16to32<<<4096, 256, 0, stream>>>(x_in,   xsm, (size_t)NN * 32);
  k_pad16to32<<<4096, 256, 0, stream>>>(eig_in, esm, (size_t)NN * 32);
  cvt_into(eattr, eab, (size_t)NE * 32);
  GEMM(0, true,  false, 32,  xsm, WiB, inproj.b, hf, nullptr, NN, 256);  // h = x@Wi+b
  GEMM(2, true,  true,  32,  esm, Wp1, pe1.b, nullptr, xb, NN, 256);     // gelu(eig@W1+b1)
  GEMM(1, true,  false, 256, xb,  Wp2, pe2.b, hf, nullptr, NN, 256);     // h += t@W2+b2

  // -------- 4 local TransformerConv blocks --------
  for (int L = 0; L < 4; ++L) {
    const Blk& B = blk[L];
    k_ln<false><<<NN/8, 256, 0, stream>>>(hf, B.n1g, B.n1b, nullptr, xb);    // xn
    GEMM(0, true,  false, 256, xb, WqB[L], B.q.b, qf, nullptr, NN, 256);
    GEMM(0, true,  false, 256, xb, WkB[L], B.k.b, kf, nullptr, NN, 256);
    GEMM(0, true,  false, 256, xb, WvB[L], B.v.b, vf, nullptr, NN, 256);
    GEMM(2, false, false, 32,  eab, WeB[L], nullptr, nullptr, ebuf, NE, 256); // edge lin
    GEMM(1, true,  false, 256, xb, WsB[L], B.skip.b, hf, nullptr, NN, 256);   // h += skip
    fill(mbuf, 0xFF800000u, (size_t)NN * 8);                                  // -inf
    fill(sbuf, 0u,          (size_t)NN * 8);
    k_edge_alpha<<<NE/8, 256, 0, stream>>>(qf, kf, ebuf, esrc, edst, alpha);
    k_segmax<<<(NE*8)/256, 256, 0, stream>>>(alpha, edst, mbuf);
    k_expsum<<<(NE*8)/256, 256, 0, stream>>>(alpha, edst, mbuf, sbuf);
    k_edge_scatter<<<NE/8, 256, 0, stream>>>(vf, ebuf, alpha, sbuf, esrc, edst, hf);
    k_ln<false><<<NN/8, 256, 0, stream>>>(hf, B.n2g, B.n2b, nullptr, xb);    // xn2
    GEMM(2, true,  true,  256,  xb,  Wf1B[L], B.ff1.b, nullptr, ffh, NN, 1024);
    GEMM(1, true,  false, 1024, ffh, Wf2B[L], B.ff2.b, hf, nullptr, NN, 256);
  }

  // -------- Perceiver aggregation --------
  cvt_into(hf, xb, (size_t)NN * 256);                      // xd (bf16)
  GEMM(0, true, false, 256, xb, WcaK, cak.b, kf, nullptr, NN, 256);  // K_ca
  GEMM(0, true, false, 256, xb, WcaV, cav.b, vf, nullptr, NN, 256);  // V_ca
  k_bcast_lat<<<2048, 256, 0, stream>>>(latents, lat, latb, (size_t)4096 * 256);

  for (int it = 0; it < 2; ++it) {
    // cross-attn
    GEMM(0, true, false, 256, latb, WcaQ, caq.b, qf, nullptr, 4096, 256);
    k_attn<<<NB * NHEAD, 256, 0, stream>>>(qf, kf, vf, attnf, NLAT, PP);
    cvt_into(attnf, attnb, (size_t)4096 * 256);
    GEMM(1, true, false, 256, attnb, WcaO, cao.b, lat, nullptr, 4096, 256);
    k_ln<true><<<4096/8, 256, 0, stream>>>(lat, cng, cnb, lat, latb);
    // self-attn
    GEMM(0, true, false, 256, latb, WsaQ, saq.b, qf,   nullptr, 4096, 256);
    GEMM(0, true, false, 256, latb, WsaK, sak.b, ksaf, nullptr, 4096, 256);
    GEMM(0, true, false, 256, latb, WsaV, sav.b, vsaf, nullptr, 4096, 256);
    k_attn<<<NB * NHEAD, 256, 0, stream>>>(qf, ksaf, vsaf, attnf, NLAT, NLAT);
    cvt_into(attnf, attnb, (size_t)4096 * 256);
    GEMM(1, true, false, 256, attnb, WsaO, sao.b, lat, nullptr, 4096, 256);
    k_ln<true><<<4096/8, 256, 0, stream>>>(lat, sng, snb, lat, latb);
    // FFN
    GEMM(2, true, true,  256,  latb, Wpf1, pff1.b, nullptr, ffh, 4096, 1024);
    GEMM(1, true, false, 1024, ffh,  Wpf2, pff2.b, lat, nullptr, 4096, 256);
    k_ln<true><<<4096/8, 256, 0, stream>>>(lat, fng, fnb, lat, latb);
  }

  // -------- readout head (M padded 64 -> 128 with zero rows) --------
  fill(gbP, 0u, (size_t)(128 * 256 * 2) / 4);
  k_mean_lat<<<NB, 256, 0, stream>>>(lat, gmean);
  k_ln<false><<<NB/8, 256, 0, stream>>>(gmean, olng, olnb, nullptr, gbP);
  GEMM(2, true, true,  256, gbP, Wo1, out1.b, nullptr, o1b, 128, 256);
  GEMM(0, true, false, 256, o1b, Wo2, out2.b, obuf, nullptr, 128, 256);
  k_norm_out<<<NB, 256, 0, stream>>>(obuf, (float*)d_out);
}
